// MultiHeadAttention_6210522710442
// MI455X (gfx1250) — compile-verified, matrix operand order checked by reading, not run on hardware
//
#include <hip/hip_runtime.h>

// Problem constants (B=2, S=2048, D=1024, H=16, HD=64, NH=B*H=32)
#define BB 2
#define SS 2048
#define DD 1024
#define HH 16
#define HD 64
#define NH 32

typedef __attribute__((ext_vector_type(16))) __bf16 v16bf;
typedef __attribute__((ext_vector_type(8)))  __bf16 v8bf;
typedef __attribute__((ext_vector_type(8)))  float  v8f;
typedef __attribute__((ext_vector_type(4)))  float  v4f;

static __device__ __forceinline__ v8f wmma_bf16(v16bf a, v16bf b, v8f c) {
  // D = A(16x32 bf16) * B(32x16 bf16) + C(16x16 f32)
  return __builtin_amdgcn_wmma_f32_16x16x32_bf16(false, a, false, b, (short)0, c,
                                                 false, false);
}

static __device__ __forceinline__ __bf16 f2bf(float f) { return (__bf16)f; }

// Fragment from two contiguous 8-element bf16 runs (A/B 16-bit layout:
// element e -> k = (e/8)*16 + half*8 + e%8; runs are contiguous in k).
static __device__ __forceinline__ v16bf frag_bf(const __bf16* p0, const __bf16* p1) {
  v8bf a = *(const v8bf*)p0;
  v8bf b = *(const v8bf*)p1;
  v16bf r;
#pragma unroll
  for (int i = 0; i < 8; i++) { r[i] = a[i]; r[8 + i] = b[i]; }
  return r;
}

// ---------------------------------------------------------------------------
// Kernel 0: one-shot fp32 -> bf16 conversion (8 elements / thread).
// Removes all per-tile redundant cvt work from the GEMM inner loops.
// ---------------------------------------------------------------------------
__global__ void cvt_bf16(const float* __restrict__ src, __bf16* __restrict__ dst) {
  const size_t i = ((size_t)blockIdx.x * blockDim.x + threadIdx.x) * 8;
  const v4f a = *(const v4f*)(src + i);
  const v4f b = *(const v4f*)(src + i + 4);
  v8bf r;
#pragma unroll
  for (int j = 0; j < 4; j++) { r[j] = f2bf(a[j]); r[4 + j] = f2bf(b[j]); }
  *(v8bf*)(dst + i) = r;
}

// ---------------------------------------------------------------------------
// Kernel 1: qkv = x @ kqv_w^T + b (all-bf16 inputs), scattered into
// head-major workspace:
//   qws/kws: [NH][S][HD]  (row contiguous in head-dim)
//   vtws   : [NH][HD][S]  (transposed: row contiguous in sequence)
// One wave computes a 32x64 block of C[M=4096][N=3072]:
//   per K-step: 2 A-frags + 4 B-frags (12 x b128) -> 8 v_wmma.
// ---------------------------------------------------------------------------
__global__ void qkv_gemm(const __bf16* __restrict__ x, const __bf16* __restrict__ w,
                         const float* __restrict__ bias, __bf16* __restrict__ qws,
                         __bf16* __restrict__ kws, __bf16* __restrict__ vtws) {
  const int lane = threadIdx.x & 31;
  const int wid  = blockIdx.x * (blockDim.x >> 5) + (threadIdx.x >> 5);
  const int NT   = (3 * DD) / 64;  // 48 column tiles
  const int mt = wid / NT, nt = wid % NT;
  const int r0 = mt * 32, j0 = nt * 64;
  const int half = lane >> 4, lm = lane & 15;
  const int ro0 = half * 8, ro1 = 16 + half * 8;

  const __bf16* arow[2];
  const __bf16* brow[4];
#pragma unroll
  for (int t = 0; t < 2; t++) arow[t] = x + (size_t)(r0 + t * 16 + lm) * DD;
#pragma unroll
  for (int t = 0; t < 4; t++) brow[t] = w + (size_t)(j0 + t * 16 + lm) * DD;

  v8f acc[2][4] = {};
  for (int kk = 0; kk < DD; kk += 32) {
    v16bf a0 = frag_bf(arow[0] + kk + ro0, arow[0] + kk + ro1);
    v16bf a1 = frag_bf(arow[1] + kk + ro0, arow[1] + kk + ro1);
#pragma unroll
    for (int t = 0; t < 4; t++) {
      v16bf b = frag_bf(brow[t] + kk + ro0, brow[t] + kk + ro1);
      acc[0][t] = wmma_bf16(a0, b, acc[0][t]);
      acc[1][t] = wmma_bf16(a1, b, acc[1][t]);
    }
  }

  float bcol[4];
#pragma unroll
  for (int t = 0; t < 4; t++) bcol[t] = bias[j0 + t * 16 + lm];

#pragma unroll
  for (int tn = 0; tn < 4; tn++) {
    const int jbase = j0 + tn * 16;         // wave-uniform scalar
    const int which = jbase >> 10;          // 0:q 1:k 2:v (never straddles)
    const int h     = (jbase & (DD - 1)) >> 6;
    const int dbase = jbase & 63;
    const int d     = dbase + lm;
#pragma unroll
    for (int tm = 0; tm < 2; tm++) {
#pragma unroll
      for (int v = 0; v < 8; v++) {
        const int r  = r0 + tm * 16 + v + half * 8;  // C row m = v + half*8
        const int b_ = r >> 11;                      // r / S
        const int s  = r & (SS - 1);
        const int n  = (b_ << 4) + h;                // head index b*H + h
        const __bf16 val = f2bf(acc[tm][tn][v] + bcol[tn]);
        if (which == 0)
          qws[(((size_t)n * SS + s) << 6) + d] = val;
        else if (which == 1)
          kws[(((size_t)n * SS + s) << 6) + d] = val;
        else
          vtws[(((size_t)n * HD + d) << 11) + s] = val;
      }
    }
  }
}

// ---------------------------------------------------------------------------
// Kernel 2: streaming-softmax attention. One wave per (head, 16-query block).
// Per 32-key step: S = Q*K^T (4 wmma), online softmax (shfl row reductions),
// P staged via LDS into A-fragment layout, O += P*V (4 wmma, V transposed).
// ---------------------------------------------------------------------------
#define ATTN_WAVES 4
__global__ void attn_kernel(const __bf16* __restrict__ qws,
                            const __bf16* __restrict__ kws,
                            const __bf16* __restrict__ vtws,
                            __bf16* __restrict__ aws) {
  __shared__ __attribute__((aligned(16))) __bf16 pbuf[ATTN_WAVES][2 * 16 * 16];
  const int lane  = threadIdx.x & 31;
  const int wslot = threadIdx.x >> 5;
  const int wid   = blockIdx.x * ATTN_WAVES + wslot;
  const int n  = wid >> 7;    // head (0..31)
  const int qb = wid & 127;   // 16-query block (0..127)
  const int half = lane >> 4, lm = lane & 15;
  const int ro0 = half * 8, ro1 = 16 + half * 8;
  __bf16* pb = &pbuf[wslot][0];

  const __bf16* qrow = qws + (((size_t)n * SS + qb * 16 + lm) << 6);
  const v16bf qf0 = frag_bf(qrow + ro0, qrow + ro1);
  const v16bf qf1 = frag_bf(qrow + 32 + ro0, qrow + 32 + ro1);

  v8f o0 = {}, o1 = {}, o2 = {}, o3 = {};
  float mrow[8], lrow[8];
#pragma unroll
  for (int v = 0; v < 8; v++) { mrow[v] = -1e30f; lrow[v] = 0.0f; }

  for (int kb = 0; kb < SS; kb += 32) {
    // --- scores: 16 queries x 32 keys, contraction over HD=64 (2 wmma each)
    v8f sacc[2];
#pragma unroll
    for (int kt = 0; kt < 2; kt++) {
      const __bf16* krow = kws + (((size_t)n * SS + kb + kt * 16 + lm) << 6);
      v16bf kf0 = frag_bf(krow + ro0, krow + ro1);
      v16bf kf1 = frag_bf(krow + 32 + ro0, krow + 32 + ro1);
      v8f c = {};
      c = wmma_bf16(qf0, kf0, c);
      c = wmma_bf16(qf1, kf1, c);
      sacc[kt] = c;
    }
    // --- online softmax (row m = v + half*8 lives on one 16-lane group)
#pragma unroll
    for (int v = 0; v < 8; v++) {
      float mx = fmaxf(sacc[0][v], sacc[1][v]);
      mx = fmaxf(mx, __shfl_xor(mx, 1, 32));
      mx = fmaxf(mx, __shfl_xor(mx, 2, 32));
      mx = fmaxf(mx, __shfl_xor(mx, 4, 32));
      mx = fmaxf(mx, __shfl_xor(mx, 8, 32));
      const float mnew  = fmaxf(mrow[v], mx);
      const float scale = __expf(mrow[v] - mnew);
      mrow[v] = mnew;
      const float p0 = __expf(sacc[0][v] - mnew);
      const float p1 = __expf(sacc[1][v] - mnew);
      sacc[0][v] = p0;
      sacc[1][v] = p1;
      float rs = p0 + p1;
      rs += __shfl_xor(rs, 1, 32);
      rs += __shfl_xor(rs, 2, 32);
      rs += __shfl_xor(rs, 4, 32);
      rs += __shfl_xor(rs, 8, 32);
      lrow[v] = lrow[v] * scale + rs;
      o0[v] *= scale; o1[v] *= scale; o2[v] *= scale; o3[v] *= scale;
    }
    // --- stage P (C-layout -> LDS [kt][16][16]) then reload in A layout
#pragma unroll
    for (int kt = 0; kt < 2; kt++)
#pragma unroll
      for (int v = 0; v < 8; v++) {
        const int m = v + half * 8;
        pb[kt * 256 + m * 16 + lm] = f2bf(sacc[kt][v]);
      }
    asm volatile("s_wait_dscnt 0x0" ::: "memory");
    const v16bf pf = frag_bf(pb + lm * 16 + half * 8, pb + 256 + lm * 16 + half * 8);
    // --- O += P * V  (V transposed: rows contiguous over keys)
#pragma unroll
    for (int nt = 0; nt < 4; nt++) {
      const __bf16* vrow = vtws + (((size_t)n * HD + nt * 16 + lm) << 11) + kb;
      v16bf vf = frag_bf(vrow + ro0, vrow + ro1);
      if (nt == 0) o0 = wmma_bf16(pf, vf, o0);
      else if (nt == 1) o1 = wmma_bf16(pf, vf, o1);
      else if (nt == 2) o2 = wmma_bf16(pf, vf, o2);
      else o3 = wmma_bf16(pf, vf, o3);
    }
  }

  // epilogue: O/l -> attn-out workspace rows (s*B + b), cols h*64 + d (bf16)
  const int b_ = n >> 4, h = n & 15;
#pragma unroll
  for (int v = 0; v < 8; v++) {
    const float inv = 1.0f / lrow[v];
    const int s = qb * 16 + v + half * 8;
    const size_t base = (((size_t)s * BB + b_) << 10) + (h << 6);
    aws[base + 0 * 16 + lm] = f2bf(o0[v] * inv);
    aws[base + 1 * 16 + lm] = f2bf(o1[v] * inv);
    aws[base + 2 * 16 + lm] = f2bf(o2[v] * inv);
    aws[base + 3 * 16 + lm] = f2bf(o3[v] * inv);
  }
}

// ---------------------------------------------------------------------------
// Kernel 3: out = attn @ out_w^T + out_b, fp32 result to [B,S,D].
// A rows are (s*B + b); one wave per 32x64 block of [4096][1024].
// ---------------------------------------------------------------------------
__global__ void out_gemm(const __bf16* __restrict__ aws, const __bf16* __restrict__ w,
                         const float* __restrict__ bias, float* __restrict__ out) {
  const int lane = threadIdx.x & 31;
  const int wid  = blockIdx.x * (blockDim.x >> 5) + (threadIdx.x >> 5);
  const int mt = wid >> 4, nt = wid & 15;  // 128 x 16 tiles
  const int r0 = mt * 32, j0 = nt * 64;
  const int half = lane >> 4, lm = lane & 15;
  const int ro0 = half * 8, ro1 = 16 + half * 8;

  const __bf16* arow[2];
  const __bf16* brow[4];
#pragma unroll
  for (int t = 0; t < 2; t++) arow[t] = aws + (size_t)(r0 + t * 16 + lm) * DD;
#pragma unroll
  for (int t = 0; t < 4; t++) brow[t] = w + (size_t)(j0 + t * 16 + lm) * DD;

  v8f acc[2][4] = {};
  for (int kk = 0; kk < DD; kk += 32) {
    v16bf a0 = frag_bf(arow[0] + kk + ro0, arow[0] + kk + ro1);
    v16bf a1 = frag_bf(arow[1] + kk + ro0, arow[1] + kk + ro1);
#pragma unroll
    for (int t = 0; t < 4; t++) {
      v16bf b = frag_bf(brow[t] + kk + ro0, brow[t] + kk + ro1);
      acc[0][t] = wmma_bf16(a0, b, acc[0][t]);
      acc[1][t] = wmma_bf16(a1, b, acc[1][t]);
    }
  }

  float bcol[4];
#pragma unroll
  for (int t = 0; t < 4; t++) bcol[t] = bias[j0 + t * 16 + lm];

#pragma unroll
  for (int tn = 0; tn < 4; tn++) {
    const int j = j0 + tn * 16 + lm;
#pragma unroll
    for (int tm = 0; tm < 2; tm++) {
#pragma unroll
      for (int v = 0; v < 8; v++) {
        const int r = r0 + tm * 16 + v + half * 8;  // r = s*B + b
        const int s = r >> 1, b_ = r & 1;
        out[(size_t)b_ * (SS * DD) + (size_t)s * DD + j] = acc[tm][tn][v] + bcol[tn];
      }
    }
  }
}

// ---------------------------------------------------------------------------
extern "C" void kernel_launch(void* const* d_in, const int* in_sizes, int n_in,
                              void* d_out, int out_size, void* d_ws, size_t ws_size,
                              hipStream_t stream) {
  const float* x     = (const float*)d_in[0];
  const float* kqv_w = (const float*)d_in[1];
  const float* kqv_b = (const float*)d_in[2];
  const float* out_w = (const float*)d_in[3];
  const float* out_b = (const float*)d_in[4];
  float* out = (float*)d_out;

  const size_t HSZ = (size_t)NH * SS * HD;  // 4M bf16 elements per head tensor
  __bf16* qws  = (__bf16*)d_ws;
  __bf16* kws  = qws + HSZ;
  __bf16* vtws = kws + HSZ;
  __bf16* aws  = vtws + HSZ;                  // [S*B][D]
  __bf16* xbf  = aws + (size_t)BB * SS * DD;  // [B*S][D]   (4M)
  __bf16* wbf  = xbf + (size_t)BB * SS * DD;  // [3D][D]    (3M)
  __bf16* owbf = wbf + (size_t)3 * DD * DD;   // [D][D]     (1M)

  // Pre-convert fp32 operands to bf16 once (8 elems/thread, 256 thr/block).
  cvt_bf16<<<(BB * SS * DD) / (256 * 8), 256, 0, stream>>>(x, xbf);
  cvt_bf16<<<(3 * DD * DD) / (256 * 8), 256, 0, stream>>>(kqv_w, wbf);
  cvt_bf16<<<(DD * DD) / (256 * 8), 256, 0, stream>>>(out_w, owbf);

  // GEMM1: (4096/32) x (3072/64) = 6144 waves, 8 waves/block
  qkv_gemm<<<768, 256, 0, stream>>>(xbf, wbf, kqv_b, qws, kws, vtws);
  // Attention: 32 heads x 128 q-blocks = 4096 waves, 4 waves/block
  attn_kernel<<<1024, 32 * ATTN_WAVES, 0, stream>>>(qws, kws, vtws, aws);
  // GEMM2: (4096/32) x (1024/64) = 2048 waves, 8 waves/block
  out_gemm<<<256, 256, 0, stream>>>(aws, owbf, out_b, out);
}
